// PEPSI_52201032516135
// MI455X (gfx1250) — matrix operand order repaired and checked by reading, not compiled, not run on hardware
//
#include <hip/hip_runtime.h>

typedef __attribute__((ext_vector_type(16))) _Float16 v16h;
typedef __attribute__((ext_vector_type(8)))  float    v8f;
typedef __attribute__((ext_vector_type(4)))  int      v4i;

#define DEV static __device__ __forceinline__

constexpr int Bn  = 8;
constexpr int Hh  = 40;
constexpr int Ww  = 40;
constexpr int Dd  = 128;
constexpr int HWn = Hh * Ww;        // 1600
constexpr int NP  = Bn * HWn;       // 12800 pixels total
constexpr int CWc = 38;             // valid-patch grid
constexpr int Cc  = CWc * CWc;      // 1444 patches per image
constexpr int K9  = 9 * Dd;         // 1152
constexpr int CP  = 1536;           // C padded to 12*128
constexpr int MT  = (HWn + 127) / 128;  // 13 M tiles (last partial)

union AFrag { v16h v; unsigned u[8]; };
union BFrag { v16h v; uint4 q[2]; _Float16 h[16]; };

DEV v8f wmma16(v16h a, v16h b, v8f c) {
  return __builtin_amdgcn_wmma_f32_16x16x32_f16(false, a, false, b, (short)0, c,
                                                false, false);
}

DEV float wave_sum(float v) {
  #pragma unroll
  for (int off = 16; off > 0; off >>= 1) v += __shfl_down(v, off, 32);
  return v;
}

// --- CDNA5 async global->LDS staging (fallback: manual load + ds_store) -----
#if __has_builtin(__builtin_amdgcn_global_load_async_to_lds_b128) && \
    __has_builtin(__builtin_amdgcn_s_wait_asynccnt)
#define USE_ASYNC 1
typedef __attribute__((address_space(1))) v4i* as1p;  // global int4*
typedef __attribute__((address_space(3))) v4i* as3p;  // LDS int4*
#else
#define USE_ASYNC 0
#endif

DEV void stage_b128(const _Float16* gsrc, _Float16* ldst) {
#if USE_ASYNC
  __builtin_amdgcn_global_load_async_to_lds_b128((as1p)gsrc, (as3p)ldst, 0, 0);
#else
  *(uint4*)ldst = *(const uint4*)gsrc;
#endif
}

DEV void stage_wait() {
#if USE_ASYNC
  __builtin_amdgcn_s_wait_asynccnt(0);
#endif
}

// ---------------------------------------------------------------------------
// Pass 0: f16 casts + per-pixel squared norms of the rounded values; also
// zero-fills the 256B zero-scratch used to feed OOB lanes of the GEMM DMAs.
// ---------------------------------------------------------------------------
__global__ __launch_bounds__(128) void prep_kernel(
    const float* __restrict__ g_in, const float* __restrict__ mask,
    _Float16* __restrict__ g16, _Float16* __restrict__ bg16,
    float* __restrict__ g2sum, float* __restrict__ bg2sum,
    _Float16* __restrict__ zbuf) {
  int r = blockIdx.x;
  int d = threadIdx.x;
  if (r == 0) zbuf[d] = (_Float16)0.f;
  float m = mask[r];
  float g = g_in[(size_t)r * Dd + d];
  _Float16 hg = (_Float16)g;
  _Float16 hb = (_Float16)(g * m);
  g16[(size_t)r * Dd + d]  = hg;
  bg16[(size_t)r * Dd + d] = hb;
  float fg = (float)hg, fb = (float)hb;
  float sg = wave_sum(fg * fg);
  float sb = wave_sum(fb * fb);
  __shared__ float rg[4], rb[4];
  int lane = d & 31, wid = d >> 5;
  if (lane == 0) { rg[wid] = sg; rb[wid] = sb; }
  __syncthreads();
  if (d == 0) {
    g2sum[r]  = rg[0] + rg[1] + rg[2] + rg[3];
    bg2sum[r] = rb[0] + rb[1] + rb[2] + rb[3];
  }
}

__global__ void wwd_kernel(const float* __restrict__ g2sum,
                           float* __restrict__ wwd) {
  int idx = blockIdx.x * 256 + threadIdx.x;
  if (idx >= NP) return;
  int b = idx / HWn, pix = idx % HWn;
  int y = pix / Ww, x = pix % Ww;
  const float* gs = g2sum + b * HWn;
  float s = 0.f;
  #pragma unroll
  for (int dy = -1; dy <= 1; ++dy)
    #pragma unroll
    for (int dx = -1; dx <= 1; ++dx) {
      int yy = y + dy, xx = x + dx;
      if ((unsigned)yy < (unsigned)Hh && (unsigned)xx < (unsigned)Ww)
        s += gs[yy * Ww + xx];
    }
  wwd[idx] = s;
}

__global__ void k1d_kernel(const float* __restrict__ bg2sum,
                           float* __restrict__ k1d) {
  int idx = blockIdx.x * 256 + threadIdx.x;
  if (idx >= Bn * Cc) return;
  int b = idx / Cc, j = idx % Cc;
  int yj = j / CWc, xj = j - yj * CWc;
  const float* gs = bg2sum + b * HWn;
  float s = 0.f;
  #pragma unroll
  for (int dy = 0; dy < 3; ++dy)
    #pragma unroll
    for (int dx = 0; dx < 3; ++dx)
      s += gs[(yj + dy) * Ww + (xj + dx)];
  k1d[idx] = s;
}

// ---------------------------------------------------------------------------
// GEMM1: CS[1600 x 1536] = PZ @ P1^T (implicit patch gather of g16 / bg16).
// 128x128 block tile, 4 waves of 64x64 (4x4 WMMA tiles), K-step 32,
// double-buffered LDS fed by async global->LDS DMA.
// ---------------------------------------------------------------------------
__global__ __launch_bounds__(128) void gemm1_kernel(
    const _Float16* __restrict__ g16, const _Float16* __restrict__ bg16,
    const _Float16* __restrict__ zbuf, float* __restrict__ CS) {
  __shared__ __align__(16) _Float16 As[2][128 * 40];  // [row][k], pitch 40
  __shared__ __align__(16) _Float16 Bs[2][128 * 40];  // [patch col][k]
  int b = blockIdx.z, mt = blockIdx.y, nt = blockIdx.x;
  int tid = threadIdx.x, lane = tid & 31, wid = tid >> 5;
  int wm = wid >> 1, wn = wid & 1;       // 2x2 waves, 64x64 each
  int half = lane >> 4, l16 = lane & 15;
  const _Float16* gB  = g16  + (size_t)b * HWn * Dd;
  const _Float16* bgB = bg16 + (size_t)b * HWn * Dd;

  v8f acc[4][4];
  v8f z = {0, 0, 0, 0, 0, 0, 0, 0};
  #pragma unroll
  for (int tm = 0; tm < 4; ++tm)
    #pragma unroll
    for (int tn = 0; tn < 4; ++tn) acc[tm][tn] = z;

  auto issue = [&](int kb, int bsel) {
    int k0 = kb * 32;
    int pp = k0 >> 7;            // patch tap 0..8
    int d0 = k0 & 127;
    int dy = pp / 3 - 1, dx = pp % 3 - 1;
    int py = pp / 3,     px = pp % 3;
    #pragma unroll
    for (int it = 0; it < 4; ++it) {
      int u = tid + it * 128;
      int rl = u >> 2, seg = u & 3;
      // A: SAME-padded windows of g16 (rows = pixels)
      int i = mt * 128 + rl;
      int iy = i / Ww, ix = i - iy * Ww;
      int yy = iy + dy, xx = ix + dx;
      const _Float16* sa = zbuf;
      if (i < HWn && (unsigned)yy < (unsigned)Hh && (unsigned)xx < (unsigned)Ww)
        sa = gB + ((size_t)(yy * Ww + xx) * Dd + d0 + seg * 8);
      stage_b128(sa, &As[bsel][rl * 40 + seg * 8]);
      // B: valid patches of bg16 (cols = patches, K-contiguous per patch)
      int j = nt * 128 + rl;
      const _Float16* sb = zbuf;
      if (j < Cc) {
        int jy = j / CWc, jx = j - jy * CWc;
        sb = bgB + ((size_t)((jy + py) * Ww + (jx + px)) * Dd + d0 + seg * 8);
      }
      stage_b128(sb, &Bs[bsel][rl * 40 + seg * 8]);
    }
  };

  constexpr int NK = K9 / 32;   // 36
  issue(0, 0);
  stage_wait();
  __syncthreads();
  for (int kb = 0; kb < NK; ++kb) {
    int cur = kb & 1;
    if (kb + 1 < NK) issue(kb + 1, cur ^ 1);
    const _Float16* Ac = As[cur];
    const _Float16* Bc = Bs[cur];
    BFrag bf[4];
    #pragma unroll
    for (int tn = 0; tn < 4; ++tn) {
      const uint4* q = (const uint4*)(Bc + (wn * 64 + tn * 16 + l16) * 40 + half * 8);
      bf[tn].q[0] = q[0]; bf[tn].q[1] = q[2];
    }
    #pragma unroll
    for (int tm = 0; tm < 4; ++tm) {
      AFrag af;
      #pragma unroll
      for (int v = 0; v < 8; ++v) {
        int K = ((v & 3) << 1) + (half << 3) + ((v >> 2) << 4);
        af.u[v] = *(const unsigned*)(Ac + (wm * 64 + tm * 16 + l16) * 40 + K);
      }
      #pragma unroll
      for (int tn = 0; tn < 4; ++tn)
        acc[tm][tn] = wmma16(af.v, bf[tn].v, acc[tm][tn]);
    }
    stage_wait();
    __syncthreads();
  }

  float* outB = CS + (size_t)b * HWn * CP;
  #pragma unroll
  for (int tm = 0; tm < 4; ++tm) {
    #pragma unroll
    for (int tn = 0; tn < 4; ++tn) {
      int jb = nt * 128 + wn * 64 + tn * 16 + l16;
      #pragma unroll
      for (int rr = 0; rr < 8; ++rr) {
        int i = mt * 128 + wm * 64 + tm * 16 + half * 8 + rr;
        if (i < HWn) outB[(size_t)i * CP + jb] = acc[tm][tn][rr];
      }
    }
  }
}

// ---------------------------------------------------------------------------
// Row stats + tanh + softmax -> CA (f16, pad cols zeroed)
// ---------------------------------------------------------------------------
__global__ __launch_bounds__(256) void stats_kernel(
    const float* __restrict__ CS, const float* __restrict__ wwd,
    const float* __restrict__ k1d, _Float16* __restrict__ CA) {
  int r = blockIdx.x;
  int b = r / HWn;
  const float* cs = CS + (size_t)r * CP;
  const float* kd = k1d + b * Cc;
  __shared__ float buf[CP];
  __shared__ float rs[8], rq[8], re[8];
  int tid = threadIdx.x, lane = tid & 31, wid = tid >> 5;
  float wv = wwd[r];
  float s = 0.f, q = 0.f;
  for (int j = tid; j < CP; j += 256) {
    float v = 0.f;
    if (j < Cc) { v = kd[j] + wv - 2.f * cs[j]; s += v; q += v * v; }
    buf[j] = v;
  }
  s = wave_sum(s); q = wave_sum(q);
  if (lane == 0) { rs[wid] = s; rq[wid] = q; }
  __syncthreads();
  float S = 0.f, Q = 0.f;
  #pragma unroll
  for (int w = 0; w < 8; ++w) { S += rs[w]; Q += rq[w]; }
  float mu  = S * (1.f / (float)Cc);
  float var = Q * (1.f / (float)Cc) - mu * mu;
  float rsd = rsqrtf(fmaxf(var, 1e-20f));
  float se = 0.f;
  for (int j = tid; j < CP; j += 256) {
    float e = 0.f;
    if (j < Cc) {
      float zz = (buf[j] - mu) * rsd;
      e = __expf(-50.f * tanhf(zz) - 50.f);   // score max = 50 -> stable
      se += e;
    }
    buf[j] = e;
  }
  se = wave_sum(se);
  if (lane == 0) re[wid] = se;
  __syncthreads();
  float SE = 0.f;
  #pragma unroll
  for (int w = 0; w < 8; ++w) SE += re[w];
  float inv = 1.f / SE;
  _Float16* ca = CA + (size_t)r * CP;
  for (int j = tid; j < CP; j += 256) ca[j] = (_Float16)(buf[j] * inv);
}

// ---------------------------------------------------------------------------
// GEMM2: contrib[1600 x 1152] = CA (1600 x 1536) @ P1 (1536 x 1152 implicit).
// N tile 128 = exactly one patch tap. B staged [k][n] (async-friendly);
// B fragments gathered as 16-bit LDS reads.
// ---------------------------------------------------------------------------
__global__ __launch_bounds__(128) void gemm2_kernel(
    const _Float16* __restrict__ CA, const _Float16* __restrict__ bg16,
    const _Float16* __restrict__ zbuf, float* __restrict__ contrib) {
  __shared__ __align__(16) _Float16 As[2][128 * 40];  // [row][k]
  __shared__ __align__(16) _Float16 Bs[2][32 * 136];  // [k][n], pitch 136
  int b = blockIdx.z, mt = blockIdx.y, nt = blockIdx.x;  // nt = patch tap
  int tid = threadIdx.x, lane = tid & 31, wid = tid >> 5;
  int wm = wid >> 1, wn = wid & 1;
  int half = lane >> 4, l16 = lane & 15;
  int py = nt / 3, px = nt % 3;
  const _Float16* bgB = bg16 + (size_t)b * HWn * Dd;
  const _Float16* caB = CA   + (size_t)b * HWn * CP;

  v8f acc[4][4];
  v8f z = {0, 0, 0, 0, 0, 0, 0, 0};
  #pragma unroll
  for (int tm = 0; tm < 4; ++tm)
    #pragma unroll
    for (int tn = 0; tn < 4; ++tn) acc[tm][tn] = z;

  auto issue = [&](int kb, int bsel) {
    int k0 = kb * 32;
    #pragma unroll
    for (int it = 0; it < 4; ++it) {
      int u = tid + it * 128;
      {  // A tile from CA (rows >= 1600 -> zero scratch)
        int rl = u >> 2, seg = u & 3;
        int i = mt * 128 + rl;
        const _Float16* sa = (i < HWn)
            ? caB + ((size_t)i * CP + k0 + seg * 8) : zbuf;
        stage_b128(sa, &As[bsel][rl * 40 + seg * 8]);
      }
      {  // B tile: rows = patches j (K), cols = d channel of tap nt
        int jl = u >> 4, seg = u & 15;
        int j = k0 + jl;
        const _Float16* sb = zbuf;
        if (j < Cc) {
          int jy = j / CWc, jx = j - jy * CWc;
          sb = bgB + ((size_t)((jy + py) * Ww + (jx + px)) * Dd + seg * 8);
        }
        stage_b128(sb, &Bs[bsel][jl * 136 + seg * 8]);
      }
    }
  };

  constexpr int NK = CP / 32;   // 48
  issue(0, 0);
  stage_wait();
  __syncthreads();
  for (int kb = 0; kb < NK; ++kb) {
    int cur = kb & 1;
    if (kb + 1 < NK) issue(kb + 1, cur ^ 1);
    const _Float16* Ac = As[cur];
    const _Float16* Bc = Bs[cur];
    BFrag bf[4];
    #pragma unroll
    for (int tn = 0; tn < 4; ++tn) {
      int nl = wn * 64 + tn * 16 + l16;
      #pragma unroll
      for (int e = 0; e < 16; ++e)
        bf[tn].h[e] = Bc[(half * 8 + (e & 7) + ((e >> 3) << 4)) * 136 + nl];
    }
    #pragma unroll
    for (int tm = 0; tm < 4; ++tm) {
      AFrag af;
      #pragma unroll
      for (int v = 0; v < 8; ++v) {
        int K = ((v & 3) << 1) + (half << 3) + ((v >> 2) << 4);
        af.u[v] = *(const unsigned*)(Ac + (wm * 64 + tm * 16 + l16) * 40 + K);
      }
      #pragma unroll
      for (int tn = 0; tn < 4; ++tn)
        acc[tm][tn] = wmma16(af.v, bf[tn].v, acc[tm][tn]);
    }
    stage_wait();
    __syncthreads();
  }

  float* outB = contrib + (size_t)b * HWn * K9;
  #pragma unroll
  for (int tm = 0; tm < 4; ++tm) {
    #pragma unroll
    for (int tn = 0; tn < 4; ++tn) {
      int nb = nt * 128 + wn * 64 + tn * 16 + l16;
      #pragma unroll
      for (int rr = 0; rr < 8; ++rr) {
        int i = mt * 128 + wm * 64 + tm * 16 + half * 8 + rr;
        if (i < HWn) outB[(size_t)i * K9 + nb] = acc[tm][tn][rr];
      }
    }
  }
}

// ---------------------------------------------------------------------------
// Fold (adjoint of patch extraction), mask blend, 1x1 conv + ELU
// ---------------------------------------------------------------------------
__global__ __launch_bounds__(128) void finalize_kernel(
    const float* __restrict__ g_in, const float* __restrict__ mask,
    const float* __restrict__ Wmat, const float* __restrict__ bias,
    const float* __restrict__ contrib, float* __restrict__ out) {
  int r = blockIdx.x, d = threadIdx.x;
  int b = r / HWn, pix = r % HWn;
  int y = pix / Ww, x = pix % Ww;
  float m = mask[r];
  float g = g_in[(size_t)r * Dd + d];
  float s = 0.f;
  #pragma unroll
  for (int p = 0; p < 9; ++p) {
    int dy = p / 3, dx = p % 3;
    int sy = y + 1 - dy, sx = x + 1 - dx;
    if ((unsigned)sy < (unsigned)Hh && (unsigned)sx < (unsigned)Ww)
      s += contrib[((size_t)b * HWn + sy * Ww + sx) * K9 + p * Dd + d];
  }
  float bgv = g * m;
  float acl = bgv + (s * (1.f / 9.f)) * (1.f - m);
  __shared__ float con[2 * Dd];
  con[d] = g;
  con[Dd + d] = acl;
  __syncthreads();
  float acc = bias[d];
  #pragma unroll 4
  for (int c = 0; c < 2 * Dd; ++c) acc = fmaf(con[c], Wmat[c * Dd + d], acc);
  out[(size_t)r * Dd + d] = (acc > 0.f) ? acc : (__expf(acc) - 1.f);
}

// ---------------------------------------------------------------------------
// Host launcher.  Workspace: zbuf 256B, g16/bg16 6.6MB, norms 0.2MB,
// CS 78.6MB f32 (reused as contrib), CA 39.3MB  => ~125 MB.
// ---------------------------------------------------------------------------
extern "C" void kernel_launch(void* const* d_in, const int* in_sizes, int n_in,
                              void* d_out, int out_size, void* d_ws,
                              size_t ws_size, hipStream_t stream) {
  (void)in_sizes; (void)n_in; (void)out_size; (void)ws_size;
  const float* g_in = (const float*)d_in[0];
  const float* mask = (const float*)d_in[1];
  const float* Wmat = (const float*)d_in[2];
  const float* bias = (const float*)d_in[3];
  float* out = (float*)d_out;

  char* p = (char*)d_ws;
  auto carve = [&](size_t bytes) -> char* {
    char* q = p;
    p += (bytes + 255) & ~(size_t)255;
    return q;
  };
  _Float16* zbuf  = (_Float16*)carve(256);
  _Float16* g16   = (_Float16*)carve((size_t)NP * Dd * 2);
  _Float16* bg16  = (_Float16*)carve((size_t)NP * Dd * 2);
  float* g2sum    = (float*)carve((size_t)NP * 4);
  float* bg2sum   = (float*)carve((size_t)NP * 4);
  float* wwd      = (float*)carve((size_t)NP * 4);
  float* k1d      = (float*)carve((size_t)Bn * Cc * 4);
  float* CS       = (float*)carve((size_t)NP * CP * 4);
  _Float16* CA    = (_Float16*)carve((size_t)NP * CP * 2);
  float* contrib  = CS;   // CS dead after stats_kernel -> reuse

  prep_kernel<<<NP, 128, 0, stream>>>(g_in, mask, g16, bg16, g2sum, bg2sum, zbuf);
  wwd_kernel<<<(NP + 255) / 256, 256, 0, stream>>>(g2sum, wwd);
  k1d_kernel<<<(Bn * Cc + 255) / 256, 256, 0, stream>>>(bg2sum, k1d);
  gemm1_kernel<<<dim3(CP / 128, MT, Bn), 128, 0, stream>>>(g16, bg16, zbuf, CS);
  stats_kernel<<<NP, 256, 0, stream>>>(CS, wwd, k1d, CA);
  gemm2_kernel<<<dim3(K9 / 128, MT, Bn), 128, 0, stream>>>(CA, bg16, zbuf, contrib);
  finalize_kernel<<<NP, 128, 0, stream>>>(g_in, mask, Wmat, bias, contrib, out);
}
